// MemoryEfficientAttention_73443940762186
// MI455X (gfx1250) — matrix operand order repaired and checked by reading, not compiled, not run on hardware
//
#include <hip/hip_runtime.h>
#include <hip/hip_bf16.h>

#define S_LEN 8192
#define EMB   512
#define HEADS 8
#define DH    64
#define WIN   1024
#define HALFW 512

typedef __bf16         bf16x16 __attribute__((ext_vector_type(16)));
typedef float          f32x8   __attribute__((ext_vector_type(8)));
typedef float          f32x4   __attribute__((ext_vector_type(4)));
typedef unsigned short u16x8   __attribute__((ext_vector_type(8)));
typedef unsigned short u16x16  __attribute__((ext_vector_type(16)));

__device__ __forceinline__ unsigned short f2bf(float f) {
  unsigned int u = __builtin_bit_cast(unsigned int, f);
  u += 0x7FFFu + ((u >> 16) & 1u);  // round-to-nearest-even
  return (unsigned short)(u >> 16);
}

__device__ __forceinline__ bf16x16 make_frag(u16x8 lo, u16x8 hi) {
  u16x16 w;
#pragma unroll
  for (int i = 0; i < 8; ++i) { w[i] = lo[i]; w[i + 8] = hi[i]; }
  return __builtin_bit_cast(bf16x16, w);
}

__device__ __forceinline__ f32x8 wmma_bf16(bf16x16 a, bf16x16 b, f32x8 c) {
  return __builtin_amdgcn_wmma_f32_16x16x32_bf16(false, a, false, b, (short)0, c,
                                                 false, false);
}

// max-reduce across the 16 lanes of each wave half (rows are independent per
// half). Uses v_permlane16_b32 (VALU cross-lane) to stay off the LDS pipe.
__device__ __forceinline__ float rowmax16(float v) {
#if __has_builtin(__builtin_amdgcn_permlane16)
  const unsigned s0[4] = {0x67452301u, 0x54761032u, 0x32107654u, 0xFEDCBA98u};
  const unsigned s1[4] = {0xEFCDAB89u, 0xDCFE98BAu, 0xBA98FEDCu, 0x76543210u};
#pragma unroll
  for (int i = 0; i < 4; ++i) {
    int u = __builtin_bit_cast(int, v);
    int w = __builtin_amdgcn_permlane16(u, u, (int)s0[i], (int)s1[i], false, false);
    v = fmaxf(v, __builtin_bit_cast(float, w));
  }
#else
  v = fmaxf(v, __shfl_xor(v, 1, 32));
  v = fmaxf(v, __shfl_xor(v, 2, 32));
  v = fmaxf(v, __shfl_xor(v, 4, 32));
  v = fmaxf(v, __shfl_xor(v, 8, 32));
#endif
  return v;
}

// ---------------------------------------------------------------- convert
__global__ __launch_bounds__(256)
void cvt_f32_bf16(const float* __restrict__ src,
                  unsigned short* __restrict__ dst, int n) {
  int i = (blockIdx.x * 256 + threadIdx.x) * 8;
  if (i < n) {
    f32x4 a = *(const f32x4*)(src + i);
    f32x4 b = *(const f32x4*)(src + i + 4);
    u16x8 o;
#pragma unroll
    for (int j = 0; j < 4; ++j) { o[j] = f2bf(a[j]); o[j + 4] = f2bf(b[j]); }
    *(u16x8*)(dst + i) = o;
  }
}

// ---------------------------------------------------------------- WMMA GEMM
// out[M,512] = oscale * (A_bf16[M,512] @ B_bf16[512,512] + bias)
// Block tile 128x128, 8 waves (4x2), wave tile 32x64 -> 8 WMMA / iteration.
// MODE 0: bf16 head-major [n/64][row][n%64]   (QKV projections)
// MODE 1: f32 row-major                        (output projection)
template <int MODE>
__global__ __launch_bounds__(256)
void gemm_bf16(const unsigned short* __restrict__ A,
               const unsigned short* __restrict__ B,
               const float* __restrict__ bias,
               unsigned short* __restrict__ outHM,
               float* __restrict__ outF, int M, float oscale) {
  const int K = EMB, N = EMB;
  __shared__ __align__(16) unsigned short As[128 * 40];  // [m][k] pitch 40
  __shared__ __align__(16) unsigned short Bs[128 * 40];  // transposed [n][k]

  const int t = threadIdx.x;
  const int lane = t & 31, wave = t >> 5;
  const int wm = wave >> 1, wn = wave & 1;  // 4x2 wave grid, 32x64 per wave
  const int r = lane & 15, h = lane >> 4;
  const int blockM = blockIdx.x * 128;
  const int blockN = blockIdx.y * 128;

  f32x8 acc[2][4];
#pragma unroll
  for (int a = 0; a < 2; ++a)
#pragma unroll
    for (int b = 0; b < 4; ++b)
#pragma unroll
      for (int j = 0; j < 8; ++j) acc[a][b][j] = 0.0f;

  const int arow = t >> 1, acol = (t & 1) * 16;   // A tile 128x32
  const int brow = t >> 3, bcol = (t & 7) * 16;   // B tile 32x128
  const unsigned short* aptr = A + (size_t)(blockM + arow) * K + acol;
  const unsigned short* bptr = B + (size_t)brow * N + blockN + bcol;

  // software pipeline: global tile lives in regs one iteration ahead
  u16x8 a0 = *(const u16x8*)aptr;
  u16x8 a1 = *(const u16x8*)(aptr + 8);
  u16x8 b0 = *(const u16x8*)bptr;
  u16x8 b1 = *(const u16x8*)(bptr + 8);

  for (int kk = 0; kk < K; kk += 32) {
    *(u16x8*)&As[arow * 40 + acol] = a0;
    *(u16x8*)&As[arow * 40 + acol + 8] = a1;
#pragma unroll
    for (int j = 0; j < 8; ++j) {
      Bs[(bcol + j) * 40 + brow] = b0[j];
      Bs[(bcol + 8 + j) * 40 + brow] = b1[j];
    }
    __syncthreads();

    if (kk + 32 < K) {
      a0 = *(const u16x8*)(aptr + kk + 32);
      a1 = *(const u16x8*)(aptr + kk + 40);
      const unsigned short* bnext = bptr + (size_t)(kk + 32) * N;
      b0 = *(const u16x8*)bnext;
      b1 = *(const u16x8*)(bnext + 8);
    }

    bf16x16 afrag[2], bfrag[4];
#pragma unroll
    for (int mt = 0; mt < 2; ++mt) {
      int rr = (wm * 32 + mt * 16 + r) * 40;
      afrag[mt] = make_frag(*(const u16x8*)&As[rr + 8 * h],
                            *(const u16x8*)&As[rr + 16 + 8 * h]);
    }
#pragma unroll
    for (int nt = 0; nt < 4; ++nt) {
      int cc = (wn * 64 + nt * 16 + r) * 40;
      bfrag[nt] = make_frag(*(const u16x8*)&Bs[cc + 8 * h],
                            *(const u16x8*)&Bs[cc + 16 + 8 * h]);
    }
#pragma unroll
    for (int mt = 0; mt < 2; ++mt)
#pragma unroll
      for (int nt = 0; nt < 4; ++nt)
        acc[mt][nt] = wmma_bf16(afrag[mt], bfrag[nt], acc[mt][nt]);
    __syncthreads();
  }

#pragma unroll
  for (int nt = 0; nt < 4; ++nt) {
    int n = blockN + wn * 64 + nt * 16 + r;
    float bvl = bias[n];
#pragma unroll
    for (int mt = 0; mt < 2; ++mt) {
      int row0 = blockM + wm * 32 + mt * 16 + 8 * h;
#pragma unroll
      for (int j = 0; j < 8; ++j) {
        float v = (acc[mt][nt][j] + bvl) * oscale;
        int row = row0 + j;
        if (MODE == 0) {
          outHM[((size_t)(n >> 6) * M + row) * DH + (n & 63)] = f2bf(v);
        } else {
          outF[(size_t)row * N + n] = v;
        }
      }
    }
  }
}

// ---------------------------------------------------------------- attention
// Sliding-window flash attention. grid=(S/128, H), block=128 (4 waves x 32 q)
// Q is pre-scaled by (1/sqrt(Dh))*log2(e); softmax runs in exp2 domain.
__global__ __launch_bounds__(128)
void attn_kernel(const unsigned short* __restrict__ qh,
                 const unsigned short* __restrict__ kh,
                 const unsigned short* __restrict__ vh,
                 unsigned short* __restrict__ attn) {
  __shared__ __align__(16) unsigned short Ks[32 * 72];      // [key][d] pitch 72
  __shared__ __align__(16) unsigned short Vt[64 * 40];      // transposed [d][key]
  __shared__ __align__(16) unsigned short Ps[4 * 32 * 32];  // per-wave P tiles

  const int t = threadIdx.x;
  const int lane = t & 31, wave = t >> 5;
  const int c = lane & 15, h = lane >> 4;
  const int head = blockIdx.y;
  const int qBlock = blockIdx.x * 128;
  const int qRow0 = qBlock + wave * 32;  // this wave: 2 strips of 16 queries
  const int i0 = qBlock & ~(WIN - 1);
  const int sk = max(0, i0 - HALFW);
  const int ek = min(S_LEN, i0 + WIN + HALFW);
  const int nk = ek - sk;  // 1536 or 2048, multiple of 32

  // all-ones B fragment: row-sum of P via WMMA (l update = l*alpha + P@1)
  bf16x16 onesf;
  {
    u16x16 w;
#pragma unroll
    for (int i = 0; i < 16; ++i) w[i] = 0x3F80;  // bf16 1.0
    onesf = __builtin_bit_cast(bf16x16, w);
  }

  // Q fragments: [strip][K-chunk]
  bf16x16 qf[2][2];
#pragma unroll
  for (int u = 0; u < 2; ++u) {
    const unsigned short* qp =
        qh + ((size_t)head * S_LEN + qRow0 + 16 * u + c) * DH;
#pragma unroll
    for (int g = 0; g < 2; ++g)
      qf[u][g] = make_frag(*(const u16x8*)(qp + 32 * g + 8 * h),
                           *(const u16x8*)(qp + 32 * g + 16 + 8 * h));
  }

  f32x8 o[2][4], lv[2];
  float m[2][8];
#pragma unroll
  for (int u = 0; u < 2; ++u)
#pragma unroll
    for (int j = 0; j < 8; ++j) { m[u][j] = -INFINITY; lv[u][j] = 0.0f; }
#pragma unroll
  for (int u = 0; u < 2; ++u)
#pragma unroll
    for (int tt = 0; tt < 4; ++tt)
#pragma unroll
      for (int j = 0; j < 8; ++j) o[u][tt][j] = 0.0f;

  const int skey = t >> 2, sdb = (t & 3) * 16;
  const unsigned short* kbase =
      kh + ((size_t)head * S_LEN + sk + skey) * DH + sdb;
  const unsigned short* vbase =
      vh + ((size_t)head * S_LEN + sk + skey) * DH + sdb;

  // pipeline: tile in regs one iteration ahead
  u16x8 k0 = *(const u16x8*)kbase;
  u16x8 k1 = *(const u16x8*)(kbase + 8);
  u16x8 v0 = *(const u16x8*)vbase;
  u16x8 v1 = *(const u16x8*)(vbase + 8);

  for (int kt = 0; kt < nk; kt += 32) {
    // stage current tile (Ks row-major, Vt transposed)
    *(u16x8*)&Ks[skey * 72 + sdb] = k0;
    *(u16x8*)&Ks[skey * 72 + sdb + 8] = k1;
#pragma unroll
    for (int j = 0; j < 8; ++j) {
      Vt[(sdb + j) * 40 + skey] = v0[j];
      Vt[(sdb + 8 + j) * 40 + skey] = v1[j];
    }
    __syncthreads();

    if (kt + 32 < nk) {
      size_t adv = (size_t)(kt + 32) * DH;
      k0 = *(const u16x8*)(kbase + adv);
      k1 = *(const u16x8*)(kbase + adv + 8);
      v0 = *(const u16x8*)(vbase + adv);
      v1 = *(const u16x8*)(vbase + adv + 8);
    }
    if (kt + 64 < nk) {
      size_t adv2 = (size_t)(kt + 64) * DH;
      __builtin_prefetch(kbase + adv2, 0, 0);
      __builtin_prefetch(vbase + adv2, 0, 0);
    }

    // K fragments loaded once, reused by both query strips
    bf16x16 kf[2][2];
#pragma unroll
    for (int n = 0; n < 2; ++n) {
      int kb = (n * 16 + c) * 72;
#pragma unroll
      for (int g = 0; g < 2; ++g)
        kf[n][g] = make_frag(*(const u16x8*)&Ks[kb + 32 * g + 8 * h],
                             *(const u16x8*)&Ks[kb + 32 * g + 16 + 8 * h]);
    }

    unsigned short* pw = &Ps[wave * 32 * 32];
    float alpha[2][8];
#pragma unroll
    for (int u = 0; u < 2; ++u) {
      // scores for strip u: two 16-key tiles (Q pre-scaled, no per-score mul)
      f32x8 sc[2];
#pragma unroll
      for (int n = 0; n < 2; ++n) {
        f32x8 a;
#pragma unroll
        for (int j = 0; j < 8; ++j) a[j] = 0.0f;
#pragma unroll
        for (int g = 0; g < 2; ++g) a = wmma_bf16(qf[u][g], kf[n][g], a);
        sc[n] = a;
      }

      // online softmax (row = 8h+j within strip); max via permlane butterfly
#pragma unroll
      for (int j = 0; j < 8; ++j) {
        float mx = rowmax16(fmaxf(sc[0][j], sc[1][j]));
        float mn = fmaxf(m[u][j], mx);
        alpha[u][j] = exp2f(m[u][j] - mn);
        m[u][j] = mn;
        sc[0][j] = exp2f(sc[0][j] - mn);
        sc[1][j] = exp2f(sc[1][j] - mn);
        lv[u][j] *= alpha[u][j];
      }
#pragma unroll
      for (int tt = 0; tt < 4; ++tt)
#pragma unroll
        for (int j = 0; j < 8; ++j) o[u][tt][j] *= alpha[u][j];

      // D-layout P -> per-wave LDS strip (rows 16u..16u+15)
#pragma unroll
      for (int n = 0; n < 2; ++n)
#pragma unroll
        for (int j = 0; j < 8; ++j)
          pw[(16 * u + 8 * h + j) * 32 + n * 16 + c] = f2bf(sc[n][j]);
    }

    // P as A-fragments (in-order DS within wave, no barrier needed)
    bf16x16 pf[2];
#pragma unroll
    for (int u = 0; u < 2; ++u) {
      int pr = (16 * u + c) * 32;
      pf[u] = make_frag(*(const u16x8*)&pw[pr + 8 * h],
                        *(const u16x8*)&pw[pr + 16 + 8 * h]);
    }

    // l += rowsum(P) via ones-matrix WMMA (replicated across columns)
    lv[0] = wmma_bf16(pf[0], onesf, lv[0]);
    lv[1] = wmma_bf16(pf[1], onesf, lv[1]);

    // O += P @ V ; V fragments loaded once, reused by both strips
#pragma unroll
    for (int tt = 0; tt < 4; ++tt) {
      int vb = (tt * 16 + c) * 40;
      bf16x16 vf = make_frag(*(const u16x8*)&Vt[vb + 8 * h],
                             *(const u16x8*)&Vt[vb + 16 + 8 * h]);
      o[0][tt] = wmma_bf16(pf[0], vf, o[0][tt]);
      o[1][tt] = wmma_bf16(pf[1], vf, o[1][tt]);
    }
    __syncthreads();
  }

  // normalize and store bf16 row-major [s][e] for the final GEMM
#pragma unroll
  for (int u = 0; u < 2; ++u)
#pragma unroll
    for (int j = 0; j < 8; ++j) {
      float inv = 1.0f / lv[u][j];
      int row = qRow0 + 16 * u + 8 * h + j;
      unsigned short* op = attn + (size_t)row * EMB + head * DH;
#pragma unroll
      for (int tt = 0; tt < 4; ++tt) op[tt * 16 + c] = f2bf(o[u][tt][j] * inv);
    }
}

// ---------------------------------------------------------------- launch
extern "C" void kernel_launch(void* const* d_in, const int* in_sizes, int n_in,
                              void* d_out, int out_size, void* d_ws,
                              size_t ws_size, hipStream_t stream) {
  const float* x  = (const float*)d_in[0];
  const float* Wq = (const float*)d_in[1];
  const float* bq = (const float*)d_in[2];
  const float* Wk = (const float*)d_in[3];
  const float* bk = (const float*)d_in[4];
  const float* Wv = (const float*)d_in[5];
  const float* bv = (const float*)d_in[6];
  const float* Wo = (const float*)d_in[7];
  const float* bo = (const float*)d_in[8];
  float* out = (float*)d_out;

  const size_t SE = (size_t)S_LEN * EMB;  // 4M elems
  const size_t EE = (size_t)EMB * EMB;    // 256K elems
  unsigned short* ws = (unsigned short*)d_ws;
  unsigned short* xb  = ws; ws += SE;
  unsigned short* wqb = ws; ws += EE;
  unsigned short* wkb = ws; ws += EE;
  unsigned short* wvb = ws; ws += EE;
  unsigned short* wob = ws; ws += EE;
  unsigned short* qhm = ws; ws += SE;
  unsigned short* khm = ws; ws += SE;
  unsigned short* vhm = ws; ws += SE;
  unsigned short* at  = ws; ws += SE;

  cvt_f32_bf16<<<(int)(SE / 2048), 256, 0, stream>>>(x, xb, (int)SE);
  cvt_f32_bf16<<<(int)(EE / 2048), 256, 0, stream>>>(Wq, wqb, (int)EE);
  cvt_f32_bf16<<<(int)(EE / 2048), 256, 0, stream>>>(Wk, wkb, (int)EE);
  cvt_f32_bf16<<<(int)(EE / 2048), 256, 0, stream>>>(Wv, wvb, (int)EE);
  cvt_f32_bf16<<<(int)(EE / 2048), 256, 0, stream>>>(Wo, wob, (int)EE);

  // fold softmax scale (1/sqrt(Dh) * log2e) into the Q projection
  const float QSCL = 0.125f * 1.4426950408889634f;
  dim3 gg(S_LEN / 128, EMB / 128);
  gemm_bf16<0><<<gg, 256, 0, stream>>>(xb, wqb, bq, qhm, nullptr, S_LEN, QSCL);
  gemm_bf16<0><<<gg, 256, 0, stream>>>(xb, wkb, bk, khm, nullptr, S_LEN, 1.0f);
  gemm_bf16<0><<<gg, 256, 0, stream>>>(xb, wvb, bv, vhm, nullptr, S_LEN, 1.0f);

  attn_kernel<<<dim3(S_LEN / 128, HEADS), 128, 0, stream>>>(qhm, khm, vhm, at);

  gemm_bf16<1><<<gg, 256, 0, stream>>>(at, wob, bo, nullptr, out, S_LEN, 1.0f);
}
